// DPMM_6828998000885
// MI455X (gfx1250) — matrix operand order repaired and checked by reading, not compiled, not run on hardware
//
#include <hip/hip_runtime.h>
#include <hip/hip_bf16.h>
#include <hip/hip_fp16.h>

// ---------------------------------------------------------------------------
// DPMM variational update for MI455X (gfx1250, wave32, WMMA).
// Dominant op A[t,i,j] = sum_n Phi[n,t] X[n,i] X[n,j]  (5.4 GFLOP) is mapped to
// v_wmma_f32_16x16x32_f16 (f16 in, f32 accumulate), 3x3 tiles of 16x16 per t,
// K = N in chunks of 32. X is stored transposed in f16 so A- and B-fragments
// are contiguous 16B loads matching the CDNA5 WMMA VGPR layouts. Phi is also
// pre-transposed to f16 (PhiT[t][n]) so the per-chunk weight vector is two
// broadcast b128 loads with the same K addressing as the X rows; the weighting
// is then 8 v_pk_mul_f16 per A tile. The WMMA loop has zero LDS traffic.
// ---------------------------------------------------------------------------

typedef _Float16 v8h  __attribute__((ext_vector_type(8)));
typedef _Float16 v16h __attribute__((ext_vector_type(16)));
typedef float    v8f  __attribute__((ext_vector_type(8)));

#define TCOMP 16
#define DDIM  8
#define HP1   41

// d_out flat offsets (reference return order)
constexpr int O_BG1   = 0;                 // 16
constexpr int O_BG2   = 16;                // 16
constexpr int O_MU    = 32;                // 128 (D,T)
constexpr int O_KAPPA = 160;               // 16
constexpr int O_NU    = 176;               // 16
constexpr int O_PSI   = 192;               // 1024 (D,D,T)
constexpr int O_WM    = 1216;              // 656 (T,41,1)
constexpr int O_WS    = 1872;              // 26896 (T,41,41)
constexpr int O_ZA    = 28768;             // 16
constexpr int O_ZB    = 28784;             // 16
constexpr int O_EA    = 28800;             // 16
constexpr int O_EB    = 28816;             // 16

// ---------------------------------------------------------------------------
__global__ void k_zero(float* p, int n) {
    int i = blockIdx.x * blockDim.x + threadIdx.x;
    if (i < n) p[i] = 0.0f;
}

// ---------------------------------------------------------------------------
// Build X = [sigmoid(inp @ W + b), 1, 0-pad] stored TRANSPOSED as f16:
// XT[c][n], c in 0..47 (41..47 zero), n in 0..Npad-1 (rows >= N zero).
__global__ void k_buildX(const float* __restrict__ inp,
                         const float* __restrict__ W,
                         const float* __restrict__ bias,
                         _Float16* __restrict__ XT, int N, int Npad) {
    __shared__ float Ws[320];
    __shared__ float bs[40];
    for (int i = threadIdx.x; i < 320; i += blockDim.x) Ws[i] = W[i];
    for (int i = threadIdx.x; i < 40; i += blockDim.x) bs[i] = bias[i];
    __syncthreads();
    int n = blockIdx.x * blockDim.x + threadIdx.x;
    if (n >= Npad) return;
    if (n >= N) {
        #pragma unroll
        for (int j = 0; j < 48; ++j) XT[(size_t)j * Npad + n] = (_Float16)0.0f;
        return;
    }
    float x[8];
    #pragma unroll
    for (int i = 0; i < 8; ++i) x[i] = inp[(size_t)n * 8 + i];
    #pragma unroll
    for (int j = 0; j < 40; ++j) {
        float z = bs[j];
        #pragma unroll
        for (int i = 0; i < 8; ++i) z += x[i] * Ws[i * 40 + j];
        float h = 1.0f / (1.0f + __expf(-z));
        XT[(size_t)j * Npad + n] = (_Float16)h;
    }
    XT[(size_t)40 * Npad + n] = (_Float16)1.0f;
    #pragma unroll
    for (int j = 41; j < 48; ++j) XT[(size_t)j * Npad + n] = (_Float16)0.0f;
}

// ---------------------------------------------------------------------------
// Transpose Phi to f16: PhiT[t][n], t in 0..15, n rows >= N zeroed.
__global__ void k_phiT(const float* __restrict__ Phi,
                       _Float16* __restrict__ PhiT, int N, int Npad) {
    int n = blockIdx.x * blockDim.x + threadIdx.x;
    if (n >= Npad) return;
    if (n >= N) {
        #pragma unroll
        for (int t = 0; t < TCOMP; ++t) PhiT[(size_t)t * Npad + n] = (_Float16)0.0f;
        return;
    }
    #pragma unroll
    for (int t = 0; t < TCOMP; ++t)
        PhiT[(size_t)t * Npad + n] = (_Float16)Phi[(size_t)n * TCOMP + t];
}

// ---------------------------------------------------------------------------
// A[t] (48x48 padded) += sum over samples of (Phi[:,t]*X)^T X via WMMA f16.
// grid = (slabs, T), block = 256 (8 waves), each wave owns 3x3 C tiles.
__global__ void k_accumA(const _Float16* __restrict__ PhiT,
                         const _Float16* __restrict__ XT,
                         int Npad, float* __restrict__ A_acc) {
    const int t      = blockIdx.y;
    const int lane   = threadIdx.x & 31;
    const int w      = threadIdx.x >> 5;
    const int laneHi = lane >> 4;        // 0: lanes 0-15, 1: lanes 16-31
    const int m0     = lane & 15;
    const int kbaseA = laneHi * 8;       // A frag K origin per ISA layout
    const int kbaseB = laneHi * 16;      // unused; B frags use kbaseA (ISA order)
    const int NC     = Npad / 32;

    v8f zero = {0.f, 0.f, 0.f, 0.f, 0.f, 0.f, 0.f, 0.f};
    v8f acc[3][3];
    #pragma unroll
    for (int a = 0; a < 3; ++a)
        #pragma unroll
        for (int b = 0; b < 3; ++b) acc[a][b] = zero;

    const _Float16* phiRow = PhiT + (size_t)t * Npad;

    int gw = blockIdx.x * 8 + w;
    int stride = gridDim.x * 8;
    for (int c = gw; c < NC; c += stride) {
        int n0 = c * 32;

        // Phi weight vector: identical per-lane K pattern to the A x-rows
        // (K = kbaseA+h / kbaseA+16+h). Broadcast loads: all lanes in a half
        // read the same 16B line of PhiT[t].
        union { v16h v; v8h h2[2]; } up;
        up.h2[0] = *(const v8h*)(phiRow + n0 + kbaseA);
        up.h2[1] = *(const v8h*)(phiRow + n0 + kbaseA + 16);
        v16h pvec = up.v;

        // B fragments: lane L, col j = tj*16+m0, K = kbaseA..+7 / kbaseA+16..+23
        v16h bfr[3];
        #pragma unroll
        for (int tj = 0; tj < 3; ++tj) {
            const _Float16* colp = XT + (size_t)(tj * 16 + m0) * Npad + n0 + kbaseA;
            union { v16h v; v8h h2[2]; } u;
            u.h2[0] = *(const v8h*)(colp);
            u.h2[1] = *(const v8h*)(colp + 16);
            bfr[tj] = u.v;
        }

        // A fragments: row i = ti*16+m0, Phi-weighted with packed f16 mult
        v16h afr[3];
        #pragma unroll
        for (int ti = 0; ti < 3; ++ti) {
            const _Float16* rowp = XT + (size_t)(ti * 16 + m0) * Npad + n0;
            union { v16h v; v8h h2[2]; } u;
            u.h2[0] = *(const v8h*)(rowp + kbaseA);        // K=kbaseA..+7
            u.h2[1] = *(const v8h*)(rowp + kbaseA + 16);   // K=kbaseA+16..+23
            afr[ti] = pvec * u.v;                          // v_pk_mul_f16 x8
        }

        #pragma unroll
        for (int ti = 0; ti < 3; ++ti)
            #pragma unroll
            for (int tj = 0; tj < 3; ++tj)
                acc[ti][tj] = __builtin_amdgcn_wmma_f32_16x16x32_f16(
                    false, afr[ti], false, bfr[tj], (short)0, acc[ti][tj],
                    false, false);
    }

    // C layout: VGPR r -> M = r + laneHi*8, N = m0
    float* At = A_acc + t * 48 * 48;
    #pragma unroll
    for (int ti = 0; ti < 3; ++ti)
        #pragma unroll
        for (int tj = 0; tj < 3; ++tj)
            #pragma unroll
            for (int r = 0; r < 8; ++r) {
                int M = ti * 16 + r + laneHi * 8;
                int Nc = tj * 16 + m0;
                atomicAdd(&At[M * 48 + Nc], acc[ti][tj][r]);
            }
}

// ---------------------------------------------------------------------------
// Fused moment reduction: sumPhi[T], S[D,D,T], t12[D,T], b[T,41].
__global__ void k_moments(const float* __restrict__ Phi,
                          const float* __restrict__ inp,
                          const float* __restrict__ y,
                          const _Float16* __restrict__ XT,
                          int N, int Npad,
                          float* sumPhiW, float* Sw, float* t12w, float* bw) {
    const int tid = threadIdx.x;  // 256
    __shared__ float sPhi[64 * 16];
    __shared__ float sIn[64 * 8];
    __shared__ float sY[64];
    __shared__ float sX[41 * 64];

    int si[4], sj[4], st[4];
    #pragma unroll
    for (int k = 0; k < 4; ++k) {
        int e = tid + 256 * k;
        si[k] = e >> 7; sj[k] = (e >> 4) & 7; st[k] = e & 15;
    }
    int bt[3] = {0, 0, 0}, bi[3] = {0, 0, 0}, bn = 0;
    #pragma unroll
    for (int k = 0; k < 3; ++k) {
        int e = tid + 256 * k;
        if (e < 656) { bt[k] = e / 41; bi[k] = e - bt[k] * 41; bn = k + 1; }
    }
    const int i12 = tid >> 4, t12i = tid & 15;

    float aS[4] = {0, 0, 0, 0}, ab[3] = {0, 0, 0}, a12 = 0.f, asp = 0.f;

    for (int base = blockIdx.x * 64; base < N; base += gridDim.x * 64) {
        int cnt = min(64, N - base);
        __syncthreads();
        for (int idx = tid; idx < cnt * 16; idx += 256) sPhi[idx] = Phi[(size_t)base * 16 + idx];
        for (int idx = tid; idx < cnt * 8;  idx += 256) sIn[idx]  = inp[(size_t)base * 8 + idx];
        for (int idx = tid; idx < cnt;      idx += 256) sY[idx]   = y[base + idx];
        for (int idx = tid; idx < 41 * 64;  idx += 256) {
            int i = idx >> 6, n = idx & 63;
            if (n < cnt) sX[idx] = (float)XT[(size_t)i * Npad + base + n];
        }
        __syncthreads();
        for (int n = 0; n < cnt; ++n) {
            float yn = sY[n];
            #pragma unroll
            for (int k = 0; k < 4; ++k)
                aS[k] += sPhi[n * 16 + st[k]] * sIn[n * 8 + si[k]] * sIn[n * 8 + sj[k]];
            #pragma unroll
            for (int k = 0; k < 3; ++k)
                if (k < bn) ab[k] += sX[bi[k] * 64 + n] * yn * sPhi[n * 16 + bt[k]];
            if (tid < 128) a12 += sIn[n * 8 + i12] * sPhi[n * 16 + t12i];
            if (tid < 16)  asp += sPhi[n * 16 + tid];
        }
    }
    #pragma unroll
    for (int k = 0; k < 4; ++k) atomicAdd(&Sw[tid + 256 * k], aS[k]);
    #pragma unroll
    for (int k = 0; k < 3; ++k) if (k < bn) atomicAdd(&bw[tid + 256 * k], ab[k]);
    if (tid < 128) atomicAdd(&t12w[tid], a12);
    if (tid < 16)  atomicAdd(&sumPhiW[tid], asp);
}

// ---------------------------------------------------------------------------
// Per-t: WS = inv(epsExp*A + zetaExp*I) (Gauss-Jordan in LDS), WMv, zetaB, t2.
__global__ void k_solve(const float* __restrict__ A_acc,
                        const float* __restrict__ b_acc,
                        const float* epsA, const float* epsB,
                        const float* zA, const float* zB,
                        float* __restrict__ out,
                        float* __restrict__ WMvW, float* __restrict__ t2w) {
    const int t = blockIdx.x;
    const int tid = threadIdx.x;  // 128
    __shared__ float Mw[41][42];
    __shared__ float Iw[41][42];
    __shared__ float fcol[41];
    __shared__ float bsh[41];
    __shared__ float wsh[41];
    __shared__ float red[128];

    const float epsExp  = epsA[t] / epsB[t];
    const float zetaExp = zA[t] / zB[t];
    const float* At = A_acc + t * 48 * 48;

    for (int e = tid; e < 41 * 41; e += 128) {
        int i = e / 41, j = e - i * 41;
        Mw[i][j] = epsExp * At[i * 48 + j] + (i == j ? zetaExp : 0.0f);
        Iw[i][j] = (i == j) ? 1.0f : 0.0f;
    }
    __syncthreads();

    for (int k = 0; k < 41; ++k) {
        float pinv = 1.0f / Mw[k][k];
        __syncthreads();
        if (tid < 82) {
            int c = tid;
            if (c < 41) Mw[k][c] *= pinv; else Iw[k][c - 41] *= pinv;
        }
        __syncthreads();
        if (tid < 41) fcol[tid] = Mw[tid][k];
        __syncthreads();
        for (int e = tid; e < 41 * 82; e += 128) {
            int r = e / 82, c = e - r * 82;
            if (r != k) {
                float f = fcol[r];
                if (c < 41) Mw[r][c] -= f * Mw[k][c];
                else        Iw[r][c - 41] -= f * Iw[k][c - 41];
            }
        }
        __syncthreads();
    }

    // WS out
    for (int e = tid; e < 41 * 41; e += 128) {
        int i = e / 41, j = e - i * 41;
        out[O_WS + t * 1681 + e] = Iw[i][j];
    }
    if (tid < 41) bsh[tid] = b_acc[t * 41 + tid];
    __syncthreads();
    if (tid < 41) {
        float s = 0.f;
        #pragma unroll
        for (int j = 0; j < 41; ++j) s += Iw[tid][j] * bsh[j];
        s *= zetaExp;
        wsh[tid] = s;
        WMvW[t * 41 + tid] = s;
        out[O_WM + t * 41 + tid] = s;
    }
    __syncthreads();

    // t2 = sum_ij A[i,j] * WS[j,i]  (raw, unregularized A)
    float part = 0.f;
    for (int e = tid; e < 41 * 41; e += 128) {
        int i = e / 41, j = e - i * 41;
        part += At[i * 48 + j] * Iw[j][i];
    }
    red[tid] = part; __syncthreads();
    for (int s = 64; s > 0; s >>= 1) { if (tid < s) red[tid] += red[tid + s]; __syncthreads(); }
    if (tid == 0) t2w[t] = red[0];
    __syncthreads();

    float p2 = 0.f;
    if (tid < 41) p2 = wsh[tid] * wsh[tid] + Iw[tid][tid];
    red[tid] = p2; __syncthreads();
    for (int s = 64; s > 0; s >>= 1) { if (tid < s) red[tid] += red[tid + s]; __syncthreads(); }
    if (tid == 0) out[O_ZB + t] = zB[t] + 0.5f * red[0];
}

// ---------------------------------------------------------------------------
// t1[t] = sum_n Phi[n,t] * (y_n - X_n . WMv_t)^2
__global__ void k_t1(const float* __restrict__ Phi,
                     const float* __restrict__ y,
                     const _Float16* __restrict__ XT,
                     const float* __restrict__ WMvW,
                     int N, int Npad, float* __restrict__ t1w) {
    __shared__ float wsh[16 * 41];
    for (int i = threadIdx.x; i < 656; i += 256) wsh[i] = WMvW[i];
    __syncthreads();
    float acc[16];
    #pragma unroll
    for (int t = 0; t < 16; ++t) acc[t] = 0.f;
    for (int n = blockIdx.x * 256 + threadIdx.x; n < N; n += gridDim.x * 256) {
        float xv[41];
        #pragma unroll
        for (int i = 0; i < 41; ++i) xv[i] = (float)XT[(size_t)i * Npad + n];
        float yn = y[n];
        #pragma unroll
        for (int t = 0; t < 16; ++t) {
            float p = 0.f;
            #pragma unroll
            for (int i = 0; i < 41; ++i) p += xv[i] * wsh[t * 41 + i];
            float e = yn - p;
            acc[t] += e * e * Phi[(size_t)n * 16 + t];
        }
    }
    int lane = threadIdx.x & 31;
    #pragma unroll
    for (int t = 0; t < 16; ++t) {
        float v = acc[t];
        #pragma unroll
        for (int off = 16; off > 0; off >>= 1) v += __shfl_down(v, off, 32);
        if (lane == 0) atomicAdd(&t1w[t], v);
    }
}

// ---------------------------------------------------------------------------
// Scalars + mu + psi.
__global__ void k_final(const float* sumPhiW, const float* t12w, const float* Sw,
                        const float* t1w, const float* t2w,
                        const float* zA, const float* epsA, const float* epsBin,
                        float* __restrict__ out) {
    __shared__ float sp[16];
    int tid = threadIdx.x;
    if (tid < 16) sp[tid] = sumPhiW[tid];
    __syncthreads();
    if (tid < 16) {
        int t = tid;
        float s = 0.f;
        for (int tt = t + 1; tt < 16; ++tt) s += sp[tt];
        out[O_BG1 + t]   = 1.0f + sp[t];
        out[O_BG2 + t]   = 1.0f + s;                  // ALPHA_DP + tail sum
        out[O_KAPPA + t] = 1000.0f + sp[t];           // KAPPA0 + sumPhi
        out[O_NU + t]    = sp[t] + 100.0f;            // sumPhi + NU0
        out[O_ZA + t]    = zA[t] + 20.5f;             // + (H+1)/2
        out[O_EA + t]    = epsA[t] + 0.5f * sp[t];
        out[O_EB + t]    = epsBin[t] + 0.5f * (t1w[t] + t2w[t]);
    }
    if (tid < 128) {
        int t = tid & 15;
        out[O_MU + tid] = t12w[tid] / (1000.0f + sp[t]);  // mu[i,t]
    }
    for (int e = tid; e < 1024; e += 256) {
        int i = e >> 7, j = (e >> 4) & 7, t = e & 15;
        float kap = 1000.0f + sp[t];
        out[O_PSI + e] = Sw[e] + (i == j ? 500.0f : 0.0f)
                         - t12w[i * 16 + t] * t12w[j * 16 + t] / kap;
    }
}

// ---------------------------------------------------------------------------
extern "C" void kernel_launch(void* const* d_in, const int* in_sizes, int n_in,
                              void* d_out, int out_size, void* d_ws, size_t ws_size,
                              hipStream_t stream) {
    const float* Phi  = (const float*)d_in[1];
    const float* inp  = (const float*)d_in[2];
    const float* yv   = (const float*)d_in[3];
    const float* W    = (const float*)d_in[4];
    const float* bias = (const float*)d_in[5];
    const float* epsA = (const float*)d_in[6];
    const float* epsB = (const float*)d_in[7];
    const float* zA   = (const float*)d_in[8];
    const float* zB   = (const float*)d_in[9];
    float* out = (float*)d_out;

    const int N = in_sizes[2] / DDIM;           // 100000
    const int Npad = ((N + 31) / 32) * 32;

    // Workspace: XT (48 x Npad f16), PhiT (16 x Npad f16), f32 accumulators.
    _Float16* XT   = (_Float16*)d_ws;
    _Float16* PhiT = XT + (size_t)48 * Npad;
    float* facc    = (float*)d_ws + (size_t)32 * Npad;  // (48+16)*Npad halves
    float* A_acc   = facc;                               // 16*48*48
    float* sumPhiW = A_acc + 16 * 48 * 48;               // 16
    float* Sw      = sumPhiW + 16;                       // 1024
    float* t12w    = Sw + 1024;                          // 128
    float* bw      = t12w + 128;                         // 656
    float* t1w     = bw + 656;                           // 16
    float* t2w     = t1w + 16;                           // 16
    float* WMvW    = t2w + 16;                           // 656
    const int ACCN = 16 * 48 * 48 + 16 + 1024 + 128 + 656 + 16 + 16 + 656;

    k_zero<<<(ACCN + 255) / 256, 256, 0, stream>>>(facc, ACCN);
    k_buildX<<<(Npad + 255) / 256, 256, 0, stream>>>(inp, W, bias, XT, N, Npad);
    k_phiT<<<(Npad + 255) / 256, 256, 0, stream>>>(Phi, PhiT, N, Npad);
    dim3 gA(40, TCOMP);
    k_accumA<<<gA, 256, 0, stream>>>(PhiT, XT, Npad, A_acc);
    k_moments<<<128, 256, 0, stream>>>(Phi, inp, yv, XT, N, Npad, sumPhiW, Sw, t12w, bw);
    k_solve<<<TCOMP, 128, 0, stream>>>(A_acc, bw, epsA, epsB, zA, zB, out, WMvW, t2w);
    k_t1<<<128, 256, 0, stream>>>(Phi, yv, XT, WMvW, N, Npad, t1w);
    k_final<<<1, 256, 0, stream>>>(sumPhiW, t12w, Sw, t1w, t2w, zA, epsA, epsB, out);
}
